// ABCFramework_17755394802208
// MI455X (gfx1250) — compile-verified, matrix operand order checked by reading, not run on hardware
//
#include <hip/hip_runtime.h>
#include <hip/hip_bf16.h>

typedef __attribute__((ext_vector_type(16))) _Float16 v16h;
typedef __attribute__((ext_vector_type(8)))  _Float16 v8h;
typedef __attribute__((ext_vector_type(8)))  float    v8f;

#define ATT_N 4096
#define ATT_D 64
#define ATT_H 4
#define ATT_B 2
#define LOG2E 1.44269504088896340736f

// ---------------------------------------------------------------------------
// Conv backbone kernels (tiny channel counts; compute-trivial)
// ---------------------------------------------------------------------------
__global__ void k_conv3x3_relu(const float* __restrict__ in,
                               const float* __restrict__ w,
                               const float* __restrict__ bias,
                               float* __restrict__ out,
                               int B, int Ci, int Co, int H, int W) {
    int idx = blockIdx.x * blockDim.x + threadIdx.x;
    int total = B * Co * H * W;
    if (idx >= total) return;
    int x  = idx % W;
    int y  = (idx / W) % H;
    int co = (idx / (W * H)) % Co;
    int b  = idx / (W * H * Co);
    float acc = bias[co];
    for (int ci = 0; ci < Ci; ++ci) {
        const float* ip = in + ((size_t)(b * Ci + ci) * H) * W;
        const float* wp = w + ((size_t)(co * Ci + ci) * 9);
        #pragma unroll
        for (int ky = 0; ky < 3; ++ky) {
            int yy = y + ky - 1;
            if (yy < 0 || yy >= H) continue;
            #pragma unroll
            for (int kx = 0; kx < 3; ++kx) {
                int xx = x + kx - 1;
                if (xx < 0 || xx >= W) continue;
                acc += ip[(size_t)yy * W + xx] * wp[ky * 3 + kx];
            }
        }
    }
    out[idx] = fmaxf(acc, 0.0f);
}

__global__ void k_maxpool2(const float* __restrict__ in, float* __restrict__ out,
                           int B, int C, int H, int W) {
    int Ho = H >> 1, Wo = W >> 1;
    int idx = blockIdx.x * blockDim.x + threadIdx.x;
    int total = B * C * Ho * Wo;
    if (idx >= total) return;
    int x = idx % Wo;
    int y = (idx / Wo) % Ho;
    int c = (idx / (Wo * Ho)) % C;
    int b = idx / (Wo * Ho * C);
    const float* ip = in + ((size_t)(b * C + c) * H) * W;
    float m = fmaxf(fmaxf(ip[(size_t)(2 * y) * W + 2 * x],     ip[(size_t)(2 * y) * W + 2 * x + 1]),
                    fmaxf(ip[(size_t)(2 * y + 1) * W + 2 * x], ip[(size_t)(2 * y + 1) * W + 2 * x + 1]));
    out[idx] = m;
}

// ---------------------------------------------------------------------------
// QKV projection: tokens [B,N,9] @ qkv_w^T.
// Q is pre-scaled by (1/sqrt(D)) * log2(e) so softmax runs in the exp2 domain.
// V is stored transposed [B,H,D,N] so AV B-fragments are contiguous per lane.
// ---------------------------------------------------------------------------
__global__ void k_qkv(const float* __restrict__ feat,   // [B,9,4096]
                      const float* __restrict__ qkv_w,  // [768,9]
                      _Float16* __restrict__ qf,        // [B,H,N,D]
                      _Float16* __restrict__ kf,        // [B,H,N,D]
                      _Float16* __restrict__ vtf) {     // [B,H,D,N]
    int idx = blockIdx.x * blockDim.x + threadIdx.x;
    int total = ATT_B * ATT_N * 768;
    if (idx >= total) return;
    int o = idx % 768;
    int n = (idx / 768) % ATT_N;
    int b = idx / (768 * ATT_N);
    float acc = 0.0f;
    #pragma unroll
    for (int c = 0; c < 9; ++c)
        acc += feat[((size_t)(b * 9 + c)) * ATT_N + n] * qkv_w[o * 9 + c];
    if (o < 256) {
        int h = o >> 6, d = o & 63;
        qf[(((size_t)(b * ATT_H + h)) * ATT_N + n) * ATT_D + d] =
            (_Float16)(acc * (0.125f * LOG2E));
    } else if (o < 512) {
        int ol = o - 256, h = ol >> 6, d = ol & 63;
        kf[(((size_t)(b * ATT_H + h)) * ATT_N + n) * ATT_D + d] = (_Float16)acc;
    } else {
        int ol = o - 512, h = ol >> 6, d = ol & 63;
        vtf[(((size_t)(b * ATT_H + h)) * ATT_D + d) * ATT_N + n] = (_Float16)acc;
    }
}

// ---------------------------------------------------------------------------
// Fragment loader (A or B of v_wmma_f32_16x16x32_f16).
// A: "row" = matrix row (lane%16). B: "row" = matrix column (lane%16).
// Halves 0..7  <- elem[row][kBase + 8*hi + i]
// Halves 8..15 <- elem[row][kBase + 16 + 8*hi + i]
// ---------------------------------------------------------------------------
__device__ __forceinline__ v16h load_frag(const _Float16* base, int row, int ld,
                                          int kBase, int hi) {
    const _Float16* p = base + (size_t)row * ld + kBase + hi * 8;
    v8h a0 = *(const v8h*)(p);
    v8h a1 = *(const v8h*)(p + 16);
    v16h r;
    #pragma unroll
    for (int i = 0; i < 8; ++i) { r[i] = a0[i]; r[i + 8] = a1[i]; }
    return r;
}

// ---------------------------------------------------------------------------
// XOR-butterfly max across the 16-lane half using v_permlane16_b32 (pure
// VALU; no LDS traffic / dscnt waits). Selector nibbles encode lane ^ m.
// ---------------------------------------------------------------------------
__device__ __forceinline__ float pl16(float v, unsigned selLo, unsigned selHi) {
    int iv = __builtin_bit_cast(int, v);
    int r  = __builtin_amdgcn_permlane16(iv, iv, (int)selLo, (int)selHi,
                                         false, false);
    return __builtin_bit_cast(float, r);
}
__device__ __forceinline__ float half_max16(float v) {
    v = fmaxf(v, pl16(v, 0x67452301u, 0xEFCDAB89u));  // ^1
    v = fmaxf(v, pl16(v, 0x54761032u, 0xDCFE98BAu));  // ^2
    v = fmaxf(v, pl16(v, 0x32107654u, 0xBA98FEDCu));  // ^4
    v = fmaxf(v, pl16(v, 0xFEDCBA98u, 0x76543210u));  // ^8
    return v;
}

// ---------------------------------------------------------------------------
// Flash attention with relative-position bias, fused roll on store.
// grid = B*H*(N/64) blocks of 128 threads (4 wave32); each wave owns 16 rows.
//
// Bias: di = rowBlk - nb is constant per col-block, so each 64x64 bias tile
// is 127 contiguous table rows; all 64 slices pre-staged in LDS (x log2e)
// and injected as the WMMA C-operand. Bias LDS reads use one per-lane base
// pointer with compile-time immediate offsets (48 - 16t + r >= 0).
// Row sums of P are computed on the matrix unit via P x ones.
// ---------------------------------------------------------------------------
__global__ void __launch_bounds__(128)
k_attn(const _Float16* __restrict__ q,      // [B,H,N,D], pre-scaled
       const _Float16* __restrict__ k,      // [B,H,N,D]
       const _Float16* __restrict__ vt,     // [B,H,D,N]
       const float*    __restrict__ table,  // [(2*160-1)^2, H]
       const int*      __restrict__ wsz,    // window_size
       float*          __restrict__ out) {  // [B,64,64,H*D]
    __shared__ _Float16 pLds[4][16][72];    // per-wave P tile, padded rows
    __shared__ float biasAll[64][128];      // 64 col-blocks x 127 diagonals

    const int lane = threadIdx.x & 31;
    const int wv   = threadIdx.x >> 5;
    const int l    = lane & 15;
    const int hi   = lane >> 4;

    const int rowBlk = blockIdx.x & 63;
    const int h      = (blockIdx.x >> 6) & 3;
    const int b      = blockIdx.x >> 8;
    const int mBase  = rowBlk * 64 + wv * 16;

    const _Float16* Q  = q  + ((size_t)(b * ATT_H + h)) * ATT_N * ATT_D;
    const _Float16* K  = k  + ((size_t)(b * ATT_H + h)) * ATT_N * ATT_D;
    const _Float16* Vt = vt + ((size_t)(b * ATT_H + h)) * ATT_D * ATT_N;

    // ---- pre-stage every col-block's 127 bias diagonals (x log2e) ----
    for (int e = threadIdx.x; e < 64 * 128; e += 128) {
        const int nb = e >> 7;
        const int u  = e & 127;
        const int rowIdx = (rowBlk - nb + 159) * 319 + 96 + u;
        biasAll[nb][u] = table[(size_t)rowIdx * ATT_H + h] * LOG2E;
    }
    __syncthreads();

    // Load Q A-fragments once (two K-chunks of the D=64 contraction)
    v16h qA0 = load_frag(Q, mBase + l, ATT_D, 0,  hi);
    v16h qA1 = load_frag(Q, mBase + l, ATT_D, 32, hi);

    // All-ones B fragment: rowsum(P) = P x ones via the matrix unit
    v16h onesB;
    #pragma unroll
    for (int i = 0; i < 16; ++i) onesB[i] = (_Float16)1.0f;

    v8f acc[4];
    float mrow[8], lrow[8];
    #pragma unroll
    for (int t = 0; t < 4; ++t) acc[t] = {};
    #pragma unroll
    for (int r = 0; r < 8; ++r) { mrow[r] = -INFINITY; lrow[r] = 0.0f; }

    const _Float16* pw = &pLds[wv][0][0];
    const int mj0 = wv * 16 + 8 * hi;       // row-in-64-tile base for this lane

    for (int nb = 0; nb < ATT_N / 64; ++nb) {
        const int nBase = nb * 64;
        // single per-lane base; all 32 bias reads become immediate offsets
        const float* bp = &biasAll[nb][mj0 - l + 15];

        // ---- S = Q K^T + bias (bias injected as WMMA C operand) ----
        v8f s[4];
        #pragma unroll
        for (int t = 0; t < 4; ++t) {
            v16h kb0 = load_frag(K, nBase + t * 16 + l, ATT_D, 0,  hi);
            v16h kb1 = load_frag(K, nBase + t * 16 + l, ATT_D, 32, hi);
            v8f c;
            #pragma unroll
            for (int r = 0; r < 8; ++r) c[r] = bp[48 - 16 * t + r];
            c = __builtin_amdgcn_wmma_f32_16x16x32_f16(false, qA0, false, kb0,
                                                       (short)0, c, false, false);
            c = __builtin_amdgcn_wmma_f32_16x16x32_f16(false, qA1, false, kb1,
                                                       (short)0, c, false, false);
            s[t] = c;
        }

        // ---- online max (permlane16 butterfly) + exp2-domain softmax ----
        float alpha[8];
        #pragma unroll
        for (int r = 0; r < 8; ++r) {
            float lm = fmaxf(fmaxf(s[0][r], s[1][r]), fmaxf(s[2][r], s[3][r]));
            lm = half_max16(lm);
            const float mn = fmaxf(mrow[r], lm);
            alpha[r] = __builtin_amdgcn_exp2f(mrow[r] - mn);
            mrow[r] = mn;
            #pragma unroll
            for (int t = 0; t < 4; ++t)
                s[t][r] = __builtin_amdgcn_exp2f(s[t][r] - mn);
        }

        // ---- C-layout f32 -> A-layout f16 via LDS (wave-private region) ----
        #pragma unroll
        for (int t = 0; t < 4; ++t)
            #pragma unroll
            for (int r = 0; r < 8; ++r)
                pLds[wv][r + 8 * hi][t * 16 + l] = (_Float16)s[t][r];
        asm volatile("s_wait_dscnt 0x0" ::: "memory");

        v16h pa0 = load_frag(pw, l, 72, 0,  hi);
        v16h pa1 = load_frag(pw, l, 72, 32, hi);

        // ---- row sums on the matrix unit: ls[r] = sum_n P[row,n] ----
        v8f ls = {};
        ls = __builtin_amdgcn_wmma_f32_16x16x32_f16(false, pa0, false, onesB,
                                                    (short)0, ls, false, false);
        ls = __builtin_amdgcn_wmma_f32_16x16x32_f16(false, pa1, false, onesB,
                                                    (short)0, ls, false, false);

        // ---- update l, rescale O accumulators ----
        #pragma unroll
        for (int r = 0; r < 8; ++r)
            lrow[r] = lrow[r] * alpha[r] + ls[r];
        #pragma unroll
        for (int t = 0; t < 4; ++t)
            #pragma unroll
            for (int r = 0; r < 8; ++r)
                acc[t][r] *= alpha[r];

        // ---- O += P V (8 WMMAs) ----
        #pragma unroll
        for (int t = 0; t < 4; ++t) {
            v16h vb0 = load_frag(Vt, t * 16 + l, ATT_N, nBase,      hi);
            v16h vb1 = load_frag(Vt, t * 16 + l, ATT_N, nBase + 32, hi);
            acc[t] = __builtin_amdgcn_wmma_f32_16x16x32_f16(false, pa0, false, vb0,
                                                            (short)0, acc[t], false, false);
            acc[t] = __builtin_amdgcn_wmma_f32_16x16x32_f16(false, pa1, false, vb1,
                                                            (short)0, acc[t], false, false);
        }
    }

    // ---- normalize (rcp + mul) + fused roll(-shift,-shift) on store ----
    const int shift = wsz[0] >> 1;
    float rinv[8];
    #pragma unroll
    for (int r = 0; r < 8; ++r) rinv[r] = __builtin_amdgcn_rcpf(lrow[r]);
    #pragma unroll
    for (int t = 0; t < 4; ++t) {
        const int d = t * 16 + l;
        #pragma unroll
        for (int r = 0; r < 8; ++r) {
            const int mg = mBase + r + 8 * hi;
            const int i0 = mg >> 6, j0 = mg & 63;
            const int ii = (i0 + 64 - shift) & 63;
            const int jj = (j0 + 64 - shift) & 63;
            out[(((size_t)(b * 64 + ii)) * 64 + jj) * (ATT_H * ATT_D) + h * ATT_D + d] =
                acc[t][r] * rinv[r];
        }
    }
}

// ---------------------------------------------------------------------------
// Host launcher
// ---------------------------------------------------------------------------
extern "C" void kernel_launch(void* const* d_in, const int* in_sizes, int n_in,
                              void* d_out, int out_size, void* d_ws, size_t ws_size,
                              hipStream_t stream) {
    (void)in_sizes; (void)n_in; (void)out_size; (void)ws_size;

    const float* x    = (const float*)d_in[0];
    const float* c1w  = (const float*)d_in[1];   const float* c1b = (const float*)d_in[2];
    const float* c2w  = (const float*)d_in[3];   const float* c2b = (const float*)d_in[4];
    const float* c3w  = (const float*)d_in[5];   const float* c3b = (const float*)d_in[6];
    const float* c4w  = (const float*)d_in[7];   const float* c4b = (const float*)d_in[8];
    const float* c5w  = (const float*)d_in[9];   const float* c5b = (const float*)d_in[10];
    const float* c6w  = (const float*)d_in[11];  const float* c6b = (const float*)d_in[12];
    const float* qkvw = (const float*)d_in[13];
    const float* btab = (const float*)d_in[14];
    const int*   wsz  = (const int*)d_in[15];
    float* out = (float*)d_out;

    // Workspace partition (256B aligned)
    char* ws = (char*)d_ws;
    size_t off = 0;
    auto alloc = [&](size_t bytes) -> char* {
        char* p = ws + off;
        off += (bytes + 255) & ~(size_t)255;
        return p;
    };
    float* t1 = (float*)alloc((size_t)2 * 3 * 256 * 256 * 4);   // conv1 out
    float* t2 = (float*)alloc((size_t)2 * 3 * 256 * 256 * 4);   // conv2 out
    float* p1 = (float*)alloc((size_t)2 * 3 * 128 * 128 * 4);   // pool1 out
    float* t3 = (float*)alloc((size_t)2 * 6 * 128 * 128 * 4);   // conv3 out
    float* t4 = (float*)alloc((size_t)2 * 6 * 128 * 128 * 4);   // conv4 out
    float* p2 = (float*)alloc((size_t)2 * 6 * 64 * 64 * 4);     // pool2 out
    float* t5 = (float*)alloc((size_t)2 * 9 * 64 * 64 * 4);     // conv5 out
    float* t6 = (float*)alloc((size_t)2 * 9 * 64 * 64 * 4);     // conv6 out
    _Float16* qf  = (_Float16*)alloc((size_t)2 * 4 * 4096 * 64 * 2);
    _Float16* kf  = (_Float16*)alloc((size_t)2 * 4 * 4096 * 64 * 2);
    _Float16* vtf = (_Float16*)alloc((size_t)2 * 4 * 64 * 4096 * 2);

    const int TB = 256;
    auto g = [](int n, int tb) { return (n + tb - 1) / tb; };

    k_conv3x3_relu<<<g(2 * 3 * 256 * 256, TB), TB, 0, stream>>>(x,  c1w, c1b, t1, 2, 1, 3, 256, 256);
    k_conv3x3_relu<<<g(2 * 3 * 256 * 256, TB), TB, 0, stream>>>(t1, c2w, c2b, t2, 2, 3, 3, 256, 256);
    k_maxpool2    <<<g(2 * 3 * 128 * 128, TB), TB, 0, stream>>>(t2, p1, 2, 3, 256, 256);
    k_conv3x3_relu<<<g(2 * 6 * 128 * 128, TB), TB, 0, stream>>>(p1, c3w, c3b, t3, 2, 3, 6, 128, 128);
    k_conv3x3_relu<<<g(2 * 6 * 128 * 128, TB), TB, 0, stream>>>(t3, c4w, c4b, t4, 2, 6, 6, 128, 128);
    k_maxpool2    <<<g(2 * 6 * 64 * 64,   TB), TB, 0, stream>>>(t4, p2, 2, 6, 128, 128);
    k_conv3x3_relu<<<g(2 * 9 * 64 * 64,   TB), TB, 0, stream>>>(p2, c5w, c5b, t5, 2, 6, 9, 64, 64);
    k_conv3x3_relu<<<g(2 * 9 * 64 * 64,   TB), TB, 0, stream>>>(t5, c6w, c6b, t6, 2, 9, 9, 64, 64);

    k_qkv<<<g(2 * 4096 * 768, TB), TB, 0, stream>>>(t6, qkvw, qf, kf, vtf);

    // B * H * (N/64) = 2 * 4 * 64 = 512 blocks, 4 wave32 each
    k_attn<<<512, 128, 0, stream>>>(qf, kf, vtf, btab, wsz, out);
}
